// SegmentUpdater_4767413698943
// MI455X (gfx1250) — matrix operand order repaired and checked by reading, not compiled, not run on hardware
//
#include <hip/hip_runtime.h>
#include <hip/hip_bf16.h>

typedef __attribute__((ext_vector_type(16))) _Float16 v16h;
typedef __attribute__((ext_vector_type(8)))  float    v8f;

#define HIDDEN 768
#define SEG_T  128

// 1/sqrt(768)
#define INV_SQRT_D 0.03608439182435161f

__global__ __launch_bounds__(256) void seg_updater_kernel(
    const float* __restrict__ utt,    // (BS, HIDDEN)
    const float* __restrict__ tok,    // (BS, T, HIDDEN)
    const float* __restrict__ mask,   // (BS, T)
    float* __restrict__ out)          // (BS, T, HIDDEN)
{
    __shared__ float s_utt[HIDDEN];
    __shared__ float s_scores[SEG_T];

    const int b      = blockIdx.x;
    const int tid    = threadIdx.x;
    const int lane   = tid & 31;
    const int wave   = tid >> 5;        // 0..7
    const int m      = lane & 15;       // token row within the wave's 16-token tile
    const int khalf  = lane >> 4;       // 0 or 1 (which K half this lane holds)
    const int tokBase = wave * 16;

    const float* uttB = utt + (size_t)b * HIDDEN;
    const float* tokB = tok + (size_t)b * SEG_T * HIDDEN;
    float*       outB = out + (size_t)b * SEG_T * HIDDEN;

    // ---- stage utterance vector into LDS (3 KB) ----
    for (int i = tid; i < HIDDEN; i += 256) s_utt[i] = uttB[i];
    __syncthreads();

    // ---- phase 1: scores via v_wmma_f32_16x16x32_f16 ----
    // A: 16 tokens x 32 K (f16). Lane layout (ISA 7.12.2, 16-bit A 16x32):
    //   lane m/khalf holds K = kc+khalf*8 .. +7  and  kc+16+khalf*8 .. +7
    // B: 32x16, all 16 columns = utterance chunk (broadcast):
    //   lane khalf holds K = kc+khalf*8 .. +7  and  kc+16+khalf*8 .. +7 (same as A).
    const float* rowp = tokB + (size_t)(tokBase + m) * HIDDEN;
    v8f c = {0.f, 0.f, 0.f, 0.f, 0.f, 0.f, 0.f, 0.f};

    for (int kc = 0; kc < HIDDEN; kc += 32) {
        const int k0 = kc + khalf * 8;
        const int k1 = kc + 16 + khalf * 8;
        const float4 a0 = *(const float4*)(rowp + k0);
        const float4 a1 = *(const float4*)(rowp + k0 + 4);
        const float4 a2 = *(const float4*)(rowp + k1);
        const float4 a3 = *(const float4*)(rowp + k1 + 4);

        const int kb = kc + khalf * 8;
        const float4 u0 = *(const float4*)(s_utt + kb);
        const float4 u1 = *(const float4*)(s_utt + kb + 4);
        const float4 u2 = *(const float4*)(s_utt + kb + 16);
        const float4 u3 = *(const float4*)(s_utt + kb + 20);

        v16h A = {(_Float16)a0.x, (_Float16)a0.y, (_Float16)a0.z, (_Float16)a0.w,
                  (_Float16)a1.x, (_Float16)a1.y, (_Float16)a1.z, (_Float16)a1.w,
                  (_Float16)a2.x, (_Float16)a2.y, (_Float16)a2.z, (_Float16)a2.w,
                  (_Float16)a3.x, (_Float16)a3.y, (_Float16)a3.z, (_Float16)a3.w};
        v16h B = {(_Float16)u0.x, (_Float16)u0.y, (_Float16)u0.z, (_Float16)u0.w,
                  (_Float16)u1.x, (_Float16)u1.y, (_Float16)u1.z, (_Float16)u1.w,
                  (_Float16)u2.x, (_Float16)u2.y, (_Float16)u2.z, (_Float16)u2.w,
                  (_Float16)u3.x, (_Float16)u3.y, (_Float16)u3.z, (_Float16)u3.w};

        c = __builtin_amdgcn_wmma_f32_16x16x32_f16(
                /*neg_a=*/false, A, /*neg_b=*/false, B,
                /*c_mod=*/(short)0, c, /*reuse_a=*/false, /*reuse_b=*/false);
    }

    // D layout: VGPR r, lanes 0-15 -> M=r; lanes 16-31 -> M=8+r. Column N is
    // irrelevant (all identical). Lane 0 / lane 16 scatter scores to LDS.
    if (lane == 0) {
        #pragma unroll
        for (int r = 0; r < 8; ++r) s_scores[tokBase + r] = c[r];
    } else if (lane == 16) {
        #pragma unroll
        for (int r = 0; r < 8; ++r) s_scores[tokBase + 8 + r] = c[r];
    }
    __syncthreads();

    // ---- phase 2a: scale + faithful mask quirk: + exp((1-mask) * -1e5) ----
    if (tid < SEG_T) {
        float s = s_scores[tid] * INV_SQRT_D
                + __expf((1.0f - mask[(size_t)b * SEG_T + tid]) * (-100000.0f));
        s_scores[tid] = s;
    }
    __syncthreads();

    // ---- phase 2b: redundant per-wave softmax stats (wave32 shuffles) ----
    float s0 = s_scores[lane];
    float s1 = s_scores[lane + 32];
    float s2 = s_scores[lane + 64];
    float s3 = s_scores[lane + 96];
    float mx = fmaxf(fmaxf(s0, s1), fmaxf(s2, s3));
    #pragma unroll
    for (int off = 16; off > 0; off >>= 1)
        mx = fmaxf(mx, __shfl_xor(mx, off, 32));
    float sum = __expf(s0 - mx) + __expf(s1 - mx) + __expf(s2 - mx) + __expf(s3 - mx);
    #pragma unroll
    for (int off = 16; off > 0; off >>= 1)
        sum += __shfl_xor(sum, off, 32);
    const float inv_sum = 1.0f / sum;

    // ---- phase 3: out = tok + p * utt, coalesced float4 streaming ----
    #pragma unroll 4
    for (int r = 0; r < 16; ++r) {
        const int t = tokBase + r;
        const float p = __expf(s_scores[t] - mx) * inv_sum;
        const float* rp = tokB + (size_t)t * HIDDEN;
        float*       op = outB + (size_t)t * HIDDEN;
        #pragma unroll
        for (int i = lane * 4; i < HIDDEN; i += 128) {
            float4 x = *(const float4*)(rp + i);
            float4 u = *(const float4*)(s_utt + i);
            float4 y = {x.x + p * u.x, x.y + p * u.y,
                        x.z + p * u.z, x.w + p * u.w};
            *(float4*)(op + i) = y;
        }
    }
}

extern "C" void kernel_launch(void* const* d_in, const int* in_sizes, int n_in,
                              void* d_out, int out_size, void* d_ws, size_t ws_size,
                              hipStream_t stream) {
    const float* utt  = (const float*)d_in[0];  // (BS, HIDDEN) f32
    const float* tok  = (const float*)d_in[1];  // (BS, T, HIDDEN) f32
    const float* mask = (const float*)d_in[2];  // (BS, T) f32
    float* out = (float*)d_out;                 // (BS, T, HIDDEN) f32

    const int bs = in_sizes[0] / HIDDEN;        // 512
    seg_updater_kernel<<<bs, 256, 0, stream>>>(utt, tok, mask, out);
}
